// Attention_local_4088808866313
// MI455X (gfx1250) — matrix operand order repaired and checked by reading, not compiled, not run on hardware
//
#include <hip/hip_runtime.h>
#include <math.h>

typedef _Float16 half_t;
typedef __attribute__((ext_vector_type(16))) _Float16 v16h;
typedef __attribute__((ext_vector_type(8)))  _Float16 v8h;
typedef __attribute__((ext_vector_type(4)))  _Float16 v4h;
typedef __attribute__((ext_vector_type(8)))  float    v8f;

#define HEADS 16
#define DH    48
#define NB    8      // batch
#define P2    256    // windows
#define KSEL  8      // top-k

__device__ __forceinline__ v16h zero_v16h() {
  v16h z;
#pragma unroll
  for (int i = 0; i < 16; ++i) z[i] = (_Float16)0.0f;
  return z;
}
__device__ __forceinline__ v8f zero_v8f() {
  v8f z;
#pragma unroll
  for (int i = 0; i < 8; ++i) z[i] = 0.0f;
  return z;
}

// ---------------------------------------------------------------------------
// Kernel 1: depthwise 5x5 conv + BN for q/k/v, emitting f16 in WMMA-friendly
// layouts:
//   q,k: [bh][window][pos][ch]   (48 contiguous halves per (window,pos) row)
//   v  : [bh][window][ch][pos]   (pre-transposed: 4 contiguous halves per ch)
// grid: (b*1024 pixels, 3 channel-chunks, 3 = q/k/v), 256 threads
// ---------------------------------------------------------------------------
__global__ __launch_bounds__(256) void qkv_conv_kernel(
    const float* __restrict__ x, const float* __restrict__ w,
    const float* __restrict__ gamma, const float* __restrict__ beta,
    const float* __restrict__ mean, const float* __restrict__ var,
    half_t* __restrict__ qw, half_t* __restrict__ kw, half_t* __restrict__ vw)
{
  const int pixb = blockIdx.x;          // b*1024 + pixel
  const int bi   = pixb >> 10;
  const int pix  = pixb & 1023;
  const int y    = pix >> 5;
  const int xx   = pix & 31;
  const int c    = blockIdx.y * 256 + threadIdx.x;   // 0..767
  const int g    = blockIdx.z;                       // 0=q 1=k 2=v

  const float* wp = w + ((size_t)(g * 768 + c)) * 25;
  float acc = 0.0f;
#pragma unroll
  for (int dy = 0; dy < 5; ++dy) {
    const int yy = y + dy - 2;
    if ((unsigned)yy < 32u) {
#pragma unroll
      for (int dx = 0; dx < 5; ++dx) {
        const int xc = xx + dx - 2;
        if ((unsigned)xc < 32u)
          acc = fmaf(x[((size_t)bi * 1024 + yy * 32 + xc) * 768 + c],
                     wp[dy * 5 + dx], acc);
      }
    }
  }
  const int gc = g * 768 + c;
  const float inv = gamma[gc] * rsqrtf(var[gc] + 1e-5f);
  const float val = acc * inv + (beta[gc] - mean[gc] * inv);

  const int head = c / DH, ch = c % DH;
  const int wy = y & 15, wx = xx & 15;
  const int pos = ((y >> 4) << 1) | (xx >> 4);       // quadrant index a*2+bb
  const size_t bhp = (size_t)((bi * HEADS + head) * P2 + (wy * 16 + wx));
  const half_t hv = (half_t)val;
  if (g == 0)      qw[(bhp * 4 + pos) * DH + ch] = hv;
  else if (g == 1) kw[(bhp * 4 + pos) * DH + ch] = hv;
  else             vw[(bhp * DH + ch) * 4 + pos] = hv;
}

// ---------------------------------------------------------------------------
// Kernel 2: top-8 indices per 256-wide row of gen_adj; one wave per row.
// ---------------------------------------------------------------------------
__global__ __launch_bounds__(256) void topk8_kernel(
    const float* __restrict__ adj, int* __restrict__ out)
{
  const int lane = threadIdx.x & 31;
  const int row  = blockIdx.x * 8 + (threadIdx.x >> 5);   // < 32768
  const float* rp = adj + (size_t)row * 256;
  float vals[8];
#pragma unroll
  for (int j = 0; j < 8; ++j) vals[j] = rp[j * 32 + lane];

#pragma unroll
  for (int t = 0; t < KSEL; ++t) {
    float bv = -3.4e38f; int bidx = 0x7fffffff;
#pragma unroll
    for (int j = 0; j < 8; ++j) {
      const int ci = j * 32 + lane;
      if (vals[j] > bv) { bv = vals[j]; bidx = ci; }
    }
#pragma unroll
    for (int off = 16; off > 0; off >>= 1) {
      const float ov = __shfl_xor(bv, off, 32);
      const int   oi = __shfl_xor(bidx, off, 32);
      if (ov > bv || (ov == bv && oi < bidx)) { bv = ov; bidx = oi; }
    }
    if (lane == 0) out[(size_t)row * KSEL + t] = bidx;
    if ((bidx & 31) == lane) vals[bidx >> 5] = -3.4e38f;  // retire winner
  }
}

// ---------------------------------------------------------------------------
// Kernel 3: gathered window attention. One wave per (b,head,window) task.
//   S(16x32) = Qpad(16x48) K^T  via 2 N-tiles x 2 K-steps of wmma f16 (K=48->64)
//   softmax rows 0..3 (fp32, shfl over lanes 0..15), transpose via LDS
//   O(16x48) = A(16x32) V      via 3 N-tiles x 1 K-step
// ---------------------------------------------------------------------------
__global__ __launch_bounds__(256) void sparse_win_attn_kernel(
    const half_t* __restrict__ qp, const half_t* __restrict__ kp,
    const half_t* __restrict__ vt, const int* __restrict__ ridx,
    float* __restrict__ out)
{
  __shared__ float sS[8][4][32];            // per-wave softmax staging
  const int lane = threadIdx.x & 31;
  const int wv   = threadIdx.x >> 5;
  const int task = blockIdx.x * 8 + wv;     // (b*16+head)*256 + p
  const int bh   = task >> 8;
  const int p    = task & 255;
  const int bi   = bh >> 4;
  const int head = bh & 15;
  const int wy = p >> 4, wx = p & 15;
  const int  lrow = lane & 15;
  const bool hi   = lane >= 16;

  const half_t* qbase = qp + ((size_t)(bh * P2 + p)) * 4 * DH;
  const int* idxp = ridx + (size_t)task * KSEL;

  // ---- Q as WMMA A-matrix, two K-steps (chans 0..31, 32..63; 48..63 pad) ----
  v16h aq0 = zero_v16h(), aq1 = zero_v16h();
  if (lrow < 4) {
    const half_t* qrow = qbase + lrow * DH;
    const int cb = hi ? 16 : 0;
    union { v16h v; v8h h[2]; } u0, u1;
    u0.h[0] = *(const v8h*)(qrow + cb);          // K = cb..cb+7
    u0.h[1] = *(const v8h*)(qrow + cb + 8);      // K = cb+8..cb+15
    aq0 = u0.v;
    u1.v = zero_v16h();
    if (!hi) { u1.h[0] = *(const v8h*)(qrow + 32); u1.h[1] = *(const v8h*)(qrow + 40); } // K = 32..47 (lanes 0-15)
    aq1 = u1.v;                                  // K >= 48 padded zero
  }

  // ---- scores: S = (Q*scale) K^T over 32 gathered keys -------------------
  v8f s[2];
#pragma unroll
  for (int nt = 0; nt < 2; ++nt) {
    const int j    = nt * 16 + lrow;             // key column for this lane
    const int widx = idxp[j >> 2];
    const int pos  = j & 3;
    const half_t* krow = kp + ((size_t)(bh * P2 + widx) * 4 + pos) * DH;
    const v16h bk0 = *(const v16h*)(krow + (hi ? 16 : 0)); // chans 0..31
    v16h bk1;
    if (hi) bk1 = zero_v16h();                   // chans 48..63 pad
    else    bk1 = *(const v16h*)(krow + 32);     // chans 32..47
    v8f acc = zero_v8f();
    acc = __builtin_amdgcn_wmma_f32_16x16x32_f16(false, aq0, false, bk0,
                                                 (short)0, acc, false, false);
    acc = __builtin_amdgcn_wmma_f32_16x16x32_f16(false, aq1, false, bk1,
                                                 (short)0, acc, false, false);
    s[nt] = acc;
  }

  // ---- softmax over 32 keys for rows 0..3 (cols striped over lanes 0..15) --
  const float scale = 0.03608439182435161f;      // 768^-0.5
  float attv[4][2];
#pragma unroll
  for (int r = 0; r < 4; ++r) {
    const float v0 = s[0][r] * scale;
    const float v1 = s[1][r] * scale;
    float m = fmaxf(v0, v1);
#pragma unroll
    for (int off = 8; off > 0; off >>= 1) m = fmaxf(m, __shfl_xor(m, off, 32));
    const float e0 = __expf(v0 - m), e1 = __expf(v1 - m);
    float ss = e0 + e1;
#pragma unroll
    for (int off = 8; off > 0; off >>= 1) ss += __shfl_xor(ss, off, 32);
    const float rs = 1.0f / ss;
    attv[r][0] = e0 * rs;
    attv[r][1] = e1 * rs;
  }
  if (lane < 16) {
#pragma unroll
    for (int r = 0; r < 4; ++r) {
      sS[wv][r][lane]      = attv[r][0];
      sS[wv][r][lane + 16] = attv[r][1];
    }
  }
  __syncthreads();

  // ---- rebuild attention in A-matrix layout (rows >=4 zero) ---------------
  v16h aat = zero_v16h();
  if (lrow < 4) {
    const int kb0 = hi ? 16 : 0;
    const int kb1 = hi ? 24 : 8;
#pragma unroll
    for (int i = 0; i < 8; ++i) {
      aat[i]     = (_Float16)sS[wv][lrow][kb0 + i];
      aat[8 + i] = (_Float16)sS[wv][lrow][kb1 + i];
    }
  }

  // ---- O = A V over 3 channel tiles; scatter with quadrant swap -----------
#pragma unroll
  for (int ct = 0; ct < 3; ++ct) {
    const int chan = ct * 16 + lrow;
    v16h bvv;
    {
      union { v16h v; v4h q[4]; } u;
#pragma unroll
      for (int k4 = 0; k4 < 4; ++k4) {
        const int kk   = (hi ? 4 : 0) + k4;     // keys keybase+4*k4 .. +3
        const int widx = idxp[kk];
        u.q[k4] = *(const v4h*)(vt + ((size_t)(bh * P2 + widx) * DH + chan) * 4);
      }
      bvv = u.v;
    }
    v8f o = zero_v8f();
    o = __builtin_amdgcn_wmma_f32_16x16x32_f16(false, aat, false, bvv,
                                               (short)0, o, false, false);
    if (lane < 16) {
#pragma unroll
      for (int r = 0; r < 4; ++r) {            // pos r = a*2+bb
        const int a = r >> 1, bb = r & 1;
        const int yo = bb * 16 + wy;           // quadrant-swapped scatter
        const int xo = a * 16 + wx;
        out[((size_t)bi * 1024 + yo * 32 + xo) * 768 + head * DH + chan] = o[r];
      }
    }
  }
}

// ---------------------------------------------------------------------------
extern "C" void kernel_launch(void* const* d_in, const int* in_sizes, int n_in,
                              void* d_out, int out_size, void* d_ws, size_t ws_size,
                              hipStream_t stream) {
  (void)in_sizes; (void)n_in; (void)out_size; (void)ws_size;
  const float* x     = (const float*)d_in[0];
  // d_in[1] = noise (unused by reference)
  const float* adj   = (const float*)d_in[2];
  const float* cw    = (const float*)d_in[3];
  const float* gamma = (const float*)d_in[4];
  const float* beta  = (const float*)d_in[5];
  const float* mean  = (const float*)d_in[6];
  const float* var   = (const float*)d_in[7];
  float* out = (float*)d_out;

  char* ws = (char*)d_ws;
  const size_t qkvBytes = (size_t)NB * HEADS * P2 * 4 * DH * sizeof(half_t); // ~12.6MB each
  half_t* qw = (half_t*)(ws);
  half_t* kw = (half_t*)(ws + qkvBytes);
  half_t* vw = (half_t*)(ws + 2 * qkvBytes);
  int*  ridx = (int*)  (ws + 3 * qkvBytes);     // 32768*8 ints

  dim3 cgrid(NB * 1024, 3, 3);
  qkv_conv_kernel<<<cgrid, 256, 0, stream>>>(x, cw, gamma, beta, mean, var,
                                             qw, kw, vw);

  const int rows = NB * HEADS * P2;             // 32768 waves
  topk8_kernel<<<rows / 8, 256, 0, stream>>>(adj, ridx);

  sparse_win_attn_kernel<<<rows / 8, 256, 0, stream>>>(qw, kw, vw, ridx, out);
}
